// emotionDetector_16149077033858
// MI455X (gfx1250) — compile-verified, matrix operand order checked by reading, not run on hardware
//
#include <hip/hip_runtime.h>
#include <hip/hip_bf16.h>
#include <math.h>

// ---------------------------------------------------------------------------
// Problem constants (from reference)
// ---------------------------------------------------------------------------
#define BB   64
#define UU   512
#define DD   768
#define HH   384
#define G3   1152          // 3*H
#define GALL 2304          // 2*3*H (fwd gates || bwd gates)
#define NC   7
#define NROW (BB*UU)       // 32768 batch-major rows (b*UU + t)

typedef _Float16 v16h __attribute__((ext_vector_type(16)));
typedef _Float16 v8h  __attribute__((ext_vector_type(8)));
typedef float    v8f  __attribute__((ext_vector_type(8)));
typedef int      v4i  __attribute__((ext_vector_type(4)));

// ---------------------------------------------------------------------------
// CDNA5 async global->LDS staging (guarded: falls back to sync ds stores)
// Builtin signature (from clang diagnostic): (v4i AS1*, v4i AS3*, imm, imm)
// ---------------------------------------------------------------------------
#if defined(__has_builtin)
#  if __has_builtin(__builtin_amdgcn_global_load_async_to_lds_b128)
#    define HAVE_ASYNC_LDS 1
#  endif
#endif
#ifndef HAVE_ASYNC_LDS
#  define HAVE_ASYNC_LDS 0
#endif

#if HAVE_ASYNC_LDS
typedef __attribute__((address_space(1))) v4i* g_v4i_ptr;
typedef __attribute__((address_space(3))) v4i* l_v4i_ptr;
#  if __has_builtin(__builtin_amdgcn_s_wait_asynccnt)
#    define WAIT_ASYNC(n) __builtin_amdgcn_s_wait_asynccnt(n)
#  else
#    define WAIT_ASYNC(n) asm volatile("s_wait_asynccnt %0" :: "i"(n))
#  endif
#else
#  define WAIT_ASYNC(n)
#endif

__device__ __forceinline__ void cp16_g2l(const _Float16* g, _Float16* l) {
#if HAVE_ASYNC_LDS
    __builtin_amdgcn_global_load_async_to_lds_b128(
        (g_v4i_ptr)g, (l_v4i_ptr)l, 0, 0);
#else
    *(v8h*)l = *(const v8h*)g;
#endif
}

// ---------------------------------------------------------------------------
// WMMA helpers
// ---------------------------------------------------------------------------
__device__ __forceinline__ v8f wmma_f16(v16h a, v16h b, v8f c) {
    // D = A(16x32 f16) * B(32x16 f16) + C(16x16 f32)
    return __builtin_amdgcn_wmma_f32_16x16x32_f16(
        /*neg_a=*/false, a, /*neg_b=*/false, b,
        /*c_mod=*/(short)0, c, /*reuse_a=*/false, /*reuse_b=*/false);
}

// Load a 16x32 f16 fragment (A-style; also used for B via W^T row loads).
// Per ISA 7.12.2: lanes 0-15 hold K=0..7 and K=16..23 of row (r0+lane),
// lanes 16-31 hold K=8..15 and K=24..31.  Two 16B loads per lane.
__device__ __forceinline__ v16h load_frag(const _Float16* __restrict__ base,
                                          int ld, int row0, int k0) {
    const int lane = threadIdx.x & 31;
    const int r    = lane & 15;
    const int hi   = lane >> 4;
    const _Float16* p = base + (size_t)(row0 + r) * (size_t)ld + k0 + hi * 8;
    v8h lo = *(const v8h*)(p);
    v8h hv = *(const v8h*)(p + 16);
    v16h f;
#pragma unroll
    for (int i = 0; i < 8; ++i) { f[i] = lo[i]; f[i + 8] = hv[i]; }
    return f;
}

// ---------------------------------------------------------------------------
// Precision-conversion kernels (fp32 -> f16 staging in workspace)
// ---------------------------------------------------------------------------
__global__ void cvt_seq_kernel(const float* __restrict__ x,
                               _Float16* __restrict__ seq16) {
    size_t i = (size_t)blockIdx.x * blockDim.x + threadIdx.x;
    const size_t n = (size_t)NROW * DD;
    if (i >= n) return;
    size_t row = i / DD, d = i - row * DD;
    // x shape (B,U,1,8,D); we need [:,:,0,0,:] -> idx = row*8*D + d
    seq16[i] = (_Float16)x[row * (size_t)(8 * DD) + d];
}

__global__ void cvt_wih_kernel(const float* __restrict__ wf,
                               const float* __restrict__ wb,
                               _Float16* __restrict__ w16) {
    size_t i = (size_t)blockIdx.x * blockDim.x + threadIdx.x;
    const size_t n = (size_t)GALL * DD;
    if (i >= n) return;
    const size_t half = (size_t)G3 * DD;
    w16[i] = (_Float16)((i < half) ? wf[i] : wb[i - half]);
}

__global__ void cvt_whh_kernel(const float* __restrict__ wf,
                               const float* __restrict__ wb,
                               _Float16* __restrict__ whh16) {
    size_t i = (size_t)blockIdx.x * blockDim.x + threadIdx.x;
    const size_t half = (size_t)G3 * HH;
    if (i >= 2 * half) return;
    whh16[i] = (_Float16)((i < half) ? wf[i] : wb[i - half]);
}

// ---------------------------------------------------------------------------
// Kernel 1: XG[32768 x 2304] = seq16 @ [w_ih_f;w_ih_b]^T + b_ih
//
// Workgroup tile 64(M) x 128(N), BK=32, double-buffered LDS staged with
// GLOBAL_LOAD_ASYNC_TO_LDS_B128.  8 waves; each wave owns a 32x32 register
// tile = 2x2 WMMA accumulators (4 wmma per stage vs 4 LDS fragment reads).
// LDS rows padded to 40 halves (80B) -> conflict-free ds_read_b128.
// ---------------------------------------------------------------------------
#define BM   64
#define BN   128
#define BK   32
#define LDP  40              // padded row pitch in halves

__global__ void gates_gemm_kernel(const _Float16* __restrict__ A,   // [32768][768]
                                  const _Float16* __restrict__ W,   // [2304][768]
                                  const float* __restrict__ bih_f,  // [1152]
                                  const float* __restrict__ bih_b,  // [1152]
                                  float* __restrict__ XG) {         // [32768][2304]
    __shared__ __align__(16) _Float16 As[2][BM * LDP];  // 2 * 5KB
    __shared__ __align__(16) _Float16 Bs[2][BN * LDP];  // 2 * 10KB

    const int tid  = threadIdx.x;
    const int wave = tid >> 5;
    const int lane = tid & 31;
    const int m_base = blockIdx.x * BM;
    const int n_base = blockIdx.y * BN;

    // wave -> 32x32 sub-tile: 2 waves over M, 4 waves over N
    const int mloc = (wave & 1) * 32;
    const int nloc = (wave >> 1) * 32;

    // staging map: 16B chunks; A: 64 rows * 4 chunks = 256 (1/thread)
    //              B: 128 rows * 4 chunks = 512 (2/thread)
    const int rowA = tid >> 2, qA = tid & 3;

    auto issue_stage = [&](int buf, int kc) {
        cp16_g2l(A + (size_t)(m_base + rowA) * DD + kc + qA * 8,
                 &As[buf][rowA * LDP + qA * 8]);
#pragma unroll
        for (int c = 0; c < 2; ++c) {
            int idx  = tid + c * 256;
            int rowB = idx >> 2, qB = idx & 3;
            cp16_g2l(W + (size_t)(n_base + rowB) * DD + kc + qB * 8,
                     &Bs[buf][rowB * LDP + qB * 8]);
        }
    };

    v8f acc00 = {}, acc01 = {}, acc10 = {}, acc11 = {};

    issue_stage(0, 0);
    const int NSTAGE = DD / BK;          // 24
    for (int s = 0; s < NSTAGE; ++s) {
        const int cur = s & 1;
        if (s + 1 < NSTAGE) {
            issue_stage(cur ^ 1, (s + 1) * BK);
            WAIT_ASYNC(3);               // drain stage s (3 async instr/wave left = s+1)
        } else {
            WAIT_ASYNC(0);
        }
        __syncthreads();

        v16h a0 = load_frag(As[cur], LDP, mloc,      0);
        v16h a1 = load_frag(As[cur], LDP, mloc + 16, 0);
        v16h b0 = load_frag(Bs[cur], LDP, nloc,      0);
        v16h b1 = load_frag(Bs[cur], LDP, nloc + 16, 0);
        acc00 = wmma_f16(a0, b0, acc00);
        acc01 = wmma_f16(a0, b1, acc01);
        acc10 = wmma_f16(a1, b0, acc10);
        acc11 = wmma_f16(a1, b1, acc11);

        __syncthreads();                 // protect buffer reuse at stage s+2
    }

    const int r  = lane & 15;
    const int hi = lane >> 4;
    const v8f* accs[4] = { &acc00, &acc01, &acc10, &acc11 };
#pragma unroll
    for (int tm = 0; tm < 2; ++tm) {
#pragma unroll
        for (int tn = 0; tn < 2; ++tn) {
            const v8f& a = *accs[tm * 2 + tn];
            const int col = n_base + nloc + tn * 16 + r;
            const float bias = (col < G3) ? bih_f[col] : bih_b[col - G3];
#pragma unroll
            for (int i = 0; i < 8; ++i) {
                const int row = m_base + mloc + tm * 16 + i + 8 * hi;
                XG[(size_t)row * GALL + col] = a[i] + bias;
            }
        }
    }
}

// ---------------------------------------------------------------------------
// Kernel 2: GRU recurrence.  4 persistent workgroups: (dir, batch-half).
// 1024 threads = 32 waves.  h (f16) and hg (f32) live in dynamic LDS.
// Per step: hg = h @ w_hh^T + b_hh via WMMA, then pointwise gate math.
// ---------------------------------------------------------------------------
#define SMEM_HG  (32 * G3 * 4)     // 147456 B : hg  [32][1152] f32
#define SMEM_H   (32 * HH * 2)     // 24576 B  : h   [32][384]  f16
#define SMEM_LEN (32 * 4)          // 128 B
#define SMEM_TOT (SMEM_HG + SMEM_H + SMEM_LEN)

__global__ void gru_scan_kernel(const _Float16* __restrict__ whh16, // [2][1152][384]
                                const float* __restrict__ XG,       // [32768][2304]
                                const float* __restrict__ bhh_f,
                                const float* __restrict__ bhh_b,
                                const float* __restrict__ h0,       // [2][64][384]
                                const int*   __restrict__ lengths,  // [64]
                                float* __restrict__ out_f,          // [64][512][384]
                                float* __restrict__ out_b) {        // [64][512][384]
    extern __shared__ char smem[];
    float*    hg   = (float*)smem;
    _Float16* hbuf = (_Float16*)(smem + SMEM_HG);
    int*      lenb = (int*)(smem + SMEM_HG + SMEM_H);

    const int dir   = blockIdx.x >> 1;
    const int halfb = blockIdx.x & 1;
    const int b0    = halfb * 32;

    const _Float16* Wd    = whh16 + (size_t)dir * G3 * HH;
    const float*    bhh   = dir ? bhh_b : bhh_f;
    const int       cbase = dir * G3;
    float*          outp  = dir ? out_b : out_f;

    // init h from h0, cache lengths
    for (int e = threadIdx.x; e < 32 * HH; e += blockDim.x) {
        int bl = e / HH, hu = e - bl * HH;
        hbuf[e] = (_Float16)h0[(size_t)dir * BB * HH + (size_t)(b0 + bl) * HH + hu];
    }
    if (threadIdx.x < 32) lenb[threadIdx.x] = lengths[b0 + threadIdx.x];
    __syncthreads();

    const int wave = threadIdx.x >> 5;
    const int lane = threadIdx.x & 31;
    const int r    = lane & 15;
    const int hi   = lane >> 4;

    for (int t = 0; t < UU; ++t) {
        // ---- hg = h @ Wd^T + bhh : 2 M-tiles x 72 N-tiles = 144 WMMA tiles
        for (int tile = wave; tile < 144; tile += 32) {
            const int m0 = (tile & 1) * 16;
            const int n0 = (tile >> 1) * 16;
            v8f acc = {};
#pragma unroll
            for (int k0 = 0; k0 < HH; k0 += 32) {
                v16h a  = load_frag(hbuf, HH, m0, k0);  // from LDS
                v16h bm = load_frag(Wd,   HH, n0, k0);  // from global (L2-resident)
                acc = wmma_f16(a, bm, acc);
            }
            const int col = n0 + r;
            const float bias = bhh[col];
#pragma unroll
            for (int i = 0; i < 8; ++i)
                hg[(m0 + i + 8 * hi) * G3 + col] = acc[i] + bias;
        }
        __syncthreads();

        // ---- pointwise gates + h update + output write
        for (int e = threadIdx.x; e < 32 * HH; e += blockDim.x) {
            const int bl = e / HH, hu = e - bl * HH;
            const int b  = b0 + bl;
            const int len = lenb[bl];
            const int tsrc = dir ? ((t < len) ? (len - 1 - t) : t) : t;

            const size_t xrow = ((size_t)b * UU + tsrc) * GALL + cbase;
            const float xr = XG[xrow + hu];
            const float xz = XG[xrow + HH + hu];
            const float xn = XG[xrow + 2 * HH + hu];
            const float hr = hg[bl * G3 + hu];
            const float hz = hg[bl * G3 + HH + hu];
            const float hn = hg[bl * G3 + 2 * HH + hu];

            const float rg = 1.0f / (1.0f + expf(-(xr + hr)));
            const float zg = 1.0f / (1.0f + expf(-(xz + hz)));
            const float ng = tanhf(xn + rg * hn);
            const float hprev = (float)hbuf[e];
            const float hnew  = (1.0f - zg) * ng + zg * hprev;

            hbuf[e] = (_Float16)hnew;
            // fwd writes at t (tsrc==t); bwd result at step t lands at time rev(t)=tsrc
            outp[((size_t)b * UU + tsrc) * HH + hu] = hnew;
        }
        __syncthreads();
    }
}

// ---------------------------------------------------------------------------
// Kernel 3: gather + FC head.  scores[j,c] = [out_f|out_b](b_j,t_j) . fc_w[c] + fc_b[c]
// ---------------------------------------------------------------------------
__global__ void fc_kernel(const float* __restrict__ out_f,
                          const float* __restrict__ out_b,
                          const int* __restrict__ bidx,
                          const int* __restrict__ tidx,
                          const float* __restrict__ fcw,  // [7][768]
                          const float* __restrict__ fcb,  // [7]
                          float* __restrict__ out, int nrows) {
    int i = blockIdx.x * blockDim.x + threadIdx.x;
    if (i >= nrows * NC) return;
    const int row = i / NC, c = i - row * NC;
    const int b = bidx[row], t = tidx[row];
    const float* pf = out_f + ((size_t)b * UU + t) * HH;
    const float* pb = out_b + ((size_t)b * UU + t) * HH;
    const float* w  = fcw + (size_t)c * (2 * HH);
    float s = fcb[c];
#pragma unroll 4
    for (int k = 0; k < HH; ++k) s += pf[k] * w[k];
#pragma unroll 4
    for (int k = 0; k < HH; ++k) s += pb[k] * w[HH + k];
    out[i] = s;
}

// ---------------------------------------------------------------------------
// Launcher
// ---------------------------------------------------------------------------
extern "C" void kernel_launch(void* const* d_in, const int* in_sizes, int n_in,
                              void* d_out, int out_size, void* d_ws, size_t ws_size,
                              hipStream_t stream) {
    const float* x      = (const float*)d_in[0];
    const float* w_ih_f = (const float*)d_in[1];
    const float* w_hh_f = (const float*)d_in[2];
    const float* b_ih_f = (const float*)d_in[3];
    const float* b_hh_f = (const float*)d_in[4];
    const float* w_ih_b = (const float*)d_in[5];
    const float* w_hh_b = (const float*)d_in[6];
    const float* b_ih_b = (const float*)d_in[7];
    const float* b_hh_b = (const float*)d_in[8];
    const float* fc_w   = (const float*)d_in[9];
    const float* fc_b   = (const float*)d_in[10];
    const float* h0     = (const float*)d_in[11];
    const int*   lengths   = (const int*)d_in[12];
    const int*   batch_idx = (const int*)d_in[13];
    const int*   time_idx  = (const int*)d_in[14];
    const int    nrows = in_sizes[13];   // total gathered rows

    // workspace layout (all offsets 256B aligned)
    char* ws = (char*)d_ws;
    const size_t OFF_SEQ16 = 0;                                   // 32768*768 f16
    const size_t OFF_W16   = OFF_SEQ16 + (size_t)NROW * DD * 2;   // 2304*768 f16
    const size_t OFF_WHH   = OFF_W16   + (size_t)GALL * DD * 2;   // 2*1152*384 f16
    const size_t OFF_XG    = OFF_WHH   + (size_t)2 * G3 * HH * 2; // 32768*2304 f32
    const size_t OFF_OUTF  = OFF_XG    + (size_t)NROW * GALL * 4; // 64*512*384 f32
    const size_t OFF_OUTB  = OFF_OUTF  + (size_t)NROW * HH * 4;

    _Float16* seq16 = (_Float16*)(ws + OFF_SEQ16);
    _Float16* w16   = (_Float16*)(ws + OFF_W16);
    _Float16* whh16 = (_Float16*)(ws + OFF_WHH);
    float*    XG    = (float*)(ws + OFF_XG);
    float*    out_f = (float*)(ws + OFF_OUTF);
    float*    out_b = (float*)(ws + OFF_OUTB);

    // 1) fp32 -> f16 staging
    {
        const size_t n = (size_t)NROW * DD;
        cvt_seq_kernel<<<dim3((unsigned)((n + 255) / 256)), dim3(256), 0, stream>>>(x, seq16);
    }
    {
        const size_t n = (size_t)GALL * DD;
        cvt_wih_kernel<<<dim3((unsigned)((n + 255) / 256)), dim3(256), 0, stream>>>(w_ih_f, w_ih_b, w16);
    }
    {
        const size_t n = (size_t)2 * G3 * HH;
        cvt_whh_kernel<<<dim3((unsigned)((n + 255) / 256)), dim3(256), 0, stream>>>(w_hh_f, w_hh_b, whh16);
    }

    // 2) big input-gate GEMM (WMMA, async-LDS double buffered): 116 GFLOP
    {
        dim3 grid(NROW / BM, GALL / BN);   // (512, 18)
        gates_gemm_kernel<<<grid, dim3(256), 0, stream>>>(seq16, w16, b_ih_f, b_ih_b, XG);
    }

    // 3) recurrence: 4 persistent WGs (dir x batch-half), LDS-resident state
    gru_scan_kernel<<<dim3(4), dim3(1024), SMEM_TOT, stream>>>(
        whh16, XG, b_hh_f, b_hh_b, h0, lengths, out_f, out_b);

    // 4) gather + FC head
    {
        int total = nrows * NC;
        fc_kernel<<<dim3((total + 255) / 256), dim3(256), 0, stream>>>(
            out_f, out_b, batch_idx, time_idx, fc_w, fc_b, (float*)d_out, nrows);
    }
}
